// SparseMoeBlock_79551384257119
// MI455X (gfx1250) — matrix operand order repaired and checked by reading, not compiled, not run on hardware
//
#include <hip/hip_runtime.h>
#include <hip/hip_bf16.h>

// ---------------------------------------------------------------------------
// Sparse MoE block (top-2 of 8 experts), H=1024, F=2048, T=16384 tokens.
// bf16 WMMA (v_wmma_f32_16x16x32_bf16), f32 accumulate.
//  - prep:  x -> bf16; w1/w3 -> [E,F,H] bf16 (transposed); w2 -> [E,H,F] bf16
//  - route: wave/token dot products + softmax/top2, per-expert token lists
//  - moe:   32-token tiles (B fragments reused across 2 M-tiles -> 2x less
//           L2 traffic), LDS-resident A (64K) and h (128K), async global->LDS
//           gather, atomic f32 scatter to output.
// Workspace assumption: ws_size >= ~136 MB.
// ---------------------------------------------------------------------------

#define H_DIM 1024
#define F_DIM 2048
#define T_TOK 16384
#define N_EXP 8
#define TILE_M 32
#define TILES_PER_EXP (T_TOK / TILE_M)   // 512

typedef __bf16 bf16_t;
typedef bf16_t v16bf __attribute__((ext_vector_type(16)));
typedef float  v8f   __attribute__((ext_vector_type(8)));
typedef int    v4i   __attribute__((ext_vector_type(4)));

union Frag { v16bf v; uint4 q[2]; };

__device__ __forceinline__ unsigned short f2bf(float f) {
    unsigned int u = __float_as_uint(f);
    unsigned int r = u + 0x7FFFu + ((u >> 16) & 1u);   // round-to-nearest-even
    return (unsigned short)(r >> 16);
}

__device__ __forceinline__ v8f vzero8() {
    v8f z;
#pragma unroll
    for (int i = 0; i < 8; ++i) z[i] = 0.0f;
    return z;
}

// A-fragment (16x32 bf16) from LDS rows [rowOff..rowOff+15].
// ISA layout: lane l<16 holds row l, K = {k0..k0+7, k0+16..k0+23};
//             lane l+16 holds row l, K = {k0+8..15, k0+24..31}.
__device__ __forceinline__ v16bf load_a_lds(const unsigned short* buf, int strideHalf,
                                            int k0, int lane, int rowOff) {
    int m    = rowOff + (lane & 15);
    int koff = (lane >> 4) << 3;                 // 0 or 8 halves
    const uint4* p = (const uint4*)(buf + m * strideHalf + k0 + koff);
    Frag f; f.q[0] = p[0]; f.q[1] = p[2];        // second chunk +16 halves
    return f.v;
}

// B-fragment (32x16 bf16) from transposed global weights: row n contiguous in K.
// ISA layout: lane n (0..15) holds col n, K = {k0..k0+7, k0+16..k0+23}; lane n+16 K = {k0+8..15, k0+24..31}.
__device__ __forceinline__ v16bf load_b_glb(const unsigned short* rowBase,
                                            int k0, int lane) {
    int koff = (lane >> 4) << 3;                 // 0 or 8 halves
    const uint4* p = (const uint4*)(rowBase + k0 + koff);
    Frag f; f.q[0] = p[0]; f.q[1] = p[2];
    return f.v;
}

#define WMMA_BF16(A, B, C) \
    __builtin_amdgcn_wmma_f32_16x16x32_bf16(false, (A), false, (B), (short)0, (C), false, false)

// ---------------------------------------------------------------------------
// Prep kernels
// ---------------------------------------------------------------------------
__global__ void cvt_x_kernel(const float* __restrict__ in,
                             unsigned short* __restrict__ out, int n4) {
    int i = blockIdx.x * 256 + threadIdx.x;
    if (i >= n4) return;
    float4 v = ((const float4*)in)[i];
    ushort4 o;
    o.x = f2bf(v.x); o.y = f2bf(v.y); o.z = f2bf(v.z); o.w = f2bf(v.w);
    ((ushort4*)out)[i] = o;
}

// in: [E][R][C] f32 ; out: [E][C][R] bf16  (32x32 LDS-tiled transpose)
__global__ void transpose_cvt_kernel(const float* __restrict__ in,
                                     unsigned short* __restrict__ out,
                                     int R, int C) {
    __shared__ float tile[32][33];
    size_t base = (size_t)blockIdx.z * R * C;
    int c0 = blockIdx.x * 32, r0 = blockIdx.y * 32;
    int tx = threadIdx.x, ty = threadIdx.y;
#pragma unroll
    for (int j = 0; j < 32; j += 8)
        tile[ty + j][tx] = in[base + (size_t)(r0 + ty + j) * C + c0 + tx];
    __syncthreads();
#pragma unroll
    for (int j = 0; j < 32; j += 8)
        out[base + (size_t)(c0 + ty + j) * R + r0 + tx] = f2bf(tile[tx][ty + j]);
}

// ---------------------------------------------------------------------------
// Router: one wave per token. logits = (x+mod) @ gate^T, softmax, top-2.
// ---------------------------------------------------------------------------
__global__ void route_kernel(const float* __restrict__ x,
                             const float* __restrict__ mod,
                             const float* __restrict__ gate,
                             float* __restrict__ logits_out,
                             int* __restrict__ counts,
                             int* __restrict__ tok_list,
                             float* __restrict__ tok_w) {
    int t    = blockIdx.x * 8 + (threadIdx.x >> 5);
    int lane = threadIdx.x & 31;
    if (t >= T_TOK) return;

    float acc[N_EXP];
#pragma unroll
    for (int e = 0; e < N_EXP; ++e) acc[e] = 0.0f;

    const float* xt = x + (size_t)t * H_DIM;
    for (int h = lane; h < H_DIM; h += 32) {
        float xv = xt[h] + mod[h];
#pragma unroll
        for (int e = 0; e < N_EXP; ++e) acc[e] += xv * gate[e * H_DIM + h];
    }
#pragma unroll
    for (int e = 0; e < N_EXP; ++e)
        for (int off = 16; off > 0; off >>= 1)
            acc[e] += __shfl_xor(acc[e], off, 32);

    if (lane == 0) {
#pragma unroll
        for (int e = 0; e < N_EXP; ++e) logits_out[(size_t)t * N_EXP + e] = acc[e];
        int i0 = 0;
#pragma unroll
        for (int e = 1; e < N_EXP; ++e) if (acc[e] > acc[i0]) i0 = e;
        int i1 = (i0 == 0) ? 1 : 0;
#pragma unroll
        for (int e = 0; e < N_EXP; ++e)
            if (e != i0 && acc[e] > acc[i1]) i1 = e;
        // softmax denominators cancel in the top-2 renormalization
        float e1 = __expf(acc[i1] - acc[i0]);
        float s  = 1.0f + e1;
        float w0 = 1.0f / s, w1v = e1 / s;

        int p0 = atomicAdd(&counts[i0], 1);
        tok_list[i0 * T_TOK + p0] = t;
        tok_w   [i0 * T_TOK + p0] = w0;
        int p1 = atomicAdd(&counts[i1], 1);
        tok_list[i1 * T_TOK + p1] = t;
        tok_w   [i1 * T_TOK + p1] = w1v;
    }
}

// ---------------------------------------------------------------------------
// MoE MLP: one block = 32 tokens (2 M-tiles) of one expert. 256 thr = 8 waves.
//   pass1: h[32,2048] = silu(x@w1) * (x@w3)   (wave -> 256 F cols)
//   pass2: out[32,1024] += rw * (h @ w2)      (wave -> 128 H cols)
// Each B fragment feeds 2 WMMAs (both M-tiles) -> halved L2 B-traffic.
// Dynamic LDS: A 32x1024 bf16 (64K) + h 32x2048 bf16 (128K) + meta (256B)
// ---------------------------------------------------------------------------
__global__ void moe_mlp_kernel(const unsigned short* __restrict__ xb,
                               const unsigned short* __restrict__ w1t,
                               const unsigned short* __restrict__ w3t,
                               const unsigned short* __restrict__ w2t,
                               const int* __restrict__ counts,
                               const int* __restrict__ tok_list,
                               const float* __restrict__ tok_w,
                               float* __restrict__ out) {
    int e    = blockIdx.x >> 9;                  // / TILES_PER_EXP
    int tile = blockIdx.x & (TILES_PER_EXP - 1);
    int base = tile * TILE_M;
    int cnt  = counts[e];
    if (base >= cnt) return;                     // uniform per block

    extern __shared__ unsigned short smem[];
    unsigned short* Abuf = smem;                           // 32*1024 halves
    unsigned short* Hbuf = smem + TILE_M * H_DIM;          // 32*2048 halves
    int*   sTok = (int*)(Hbuf + TILE_M * F_DIM);           // 32 ints
    float* sW   = (float*)(sTok + TILE_M);                 // 32 floats

    int tid  = threadIdx.x;
    int wv   = tid >> 5;
    int lane = tid & 31;

    if (tid < TILE_M) {
        int idx = base + tid;
        if (idx < cnt) {
            sTok[tid] = tok_list[e * T_TOK + idx];
            sW[tid]   = tok_w   [e * T_TOK + idx];
        } else {                                 // pad: valid row, zero weight
            sTok[tid] = tok_list[e * T_TOK + base];
            sW[tid]   = 0.0f;
        }
    }
    __syncthreads();

    // ---- gather 32 bf16 token rows into LDS (async global->LDS if present)
#if defined(__AMDGCN__) && __has_builtin(__builtin_amdgcn_global_load_async_to_lds_b128)
    typedef __attribute__((address_space(1))) v4i* gas_ptr;   // global int4*
    typedef __attribute__((address_space(3))) v4i* las_ptr;   // LDS int4*
    for (int i = tid; i < TILE_M * H_DIM / 8; i += 256) {
        int m = i >> 7;                          // 128 uint4 per row
        int c = i & 127;
        gas_ptr g = (gas_ptr)(unsigned long long)(xb + (size_t)sTok[m] * H_DIM + c * 8);
        las_ptr l = (las_ptr)(unsigned int)(unsigned long long)(Abuf + (size_t)i * 8);
        __builtin_amdgcn_global_load_async_to_lds_b128(g, l, 0, 0);
    }
#if __has_builtin(__builtin_amdgcn_s_wait_asynccnt)
    __builtin_amdgcn_s_wait_asynccnt(0);
#else
    asm volatile("s_wait_asynccnt 0x0" ::: "memory");
#endif
#else
    for (int i = tid; i < TILE_M * H_DIM / 8; i += 256) {
        int m = i >> 7;
        int c = i & 127;
        ((uint4*)Abuf)[i] = ((const uint4*)(xb + (size_t)sTok[m] * H_DIM))[c];
    }
#endif
    __syncthreads();

    const unsigned short* w1e = w1t + (size_t)e * F_DIM * H_DIM;  // [F][H]
    const unsigned short* w3e = w3t + (size_t)e * F_DIM * H_DIM;  // [F][H]
    const unsigned short* w2e = w2t + (size_t)e * H_DIM * F_DIM;  // [H][F]

    // ------------------- pass 1: h = silu(x@w1) * (x@w3) -------------------
    int fw = wv * 256;                           // wave's 256 F columns
    for (int g = 0; g < 8; ++g) {                // 8 groups x 2 N-tiles
        int fg = fw + g * 32;
        v8f a1[2][2], a3[2][2];                  // [m-tile][n-tile]
#pragma unroll
        for (int mt = 0; mt < 2; ++mt)
#pragma unroll
            for (int nt = 0; nt < 2; ++nt) { a1[mt][nt] = vzero8(); a3[mt][nt] = vzero8(); }

        for (int k0 = 0; k0 < H_DIM; k0 += 32) {
            v16bf A0 = load_a_lds(Abuf, H_DIM, k0, lane, 0);
            v16bf A1 = load_a_lds(Abuf, H_DIM, k0, lane, 16);
#pragma unroll
            for (int nt = 0; nt < 2; ++nt) {
                int n = fg + nt * 16 + (lane & 15);
                v16bf B1 = load_b_glb(w1e + (size_t)n * H_DIM, k0, lane);
                a1[0][nt] = WMMA_BF16(A0, B1, a1[0][nt]);
                a1[1][nt] = WMMA_BF16(A1, B1, a1[1][nt]);
                v16bf B3 = load_b_glb(w3e + (size_t)n * H_DIM, k0, lane);
                a3[0][nt] = WMMA_BF16(A0, B3, a3[0][nt]);
                a3[1][nt] = WMMA_BF16(A1, B3, a3[1][nt]);
            }
        }
#pragma unroll
        for (int mt = 0; mt < 2; ++mt) {
            int mbase = mt * 16 + ((lane >> 4) << 3);
#pragma unroll
            for (int nt = 0; nt < 2; ++nt) {
                int col = fg + nt * 16 + (lane & 15);
#pragma unroll
                for (int r = 0; r < 8; ++r) {
                    float z  = a1[mt][nt][r];
                    float hv = (z / (1.0f + __expf(-z))) * a3[mt][nt][r];  // silu*gate
                    Hbuf[(mbase + r) * F_DIM + col] = f2bf(hv);
                }
            }
        }
    }
    __syncthreads();

    // ------------------- pass 2: out += rw * (h @ w2) ----------------------
    int hb = wv * 128;                           // wave's 128 H columns
    for (int g = 0; g < 2; ++g) {                // 2 groups x 4 N-tiles
        int hg = hb + g * 64;
        v8f acc[2][4];
#pragma unroll
        for (int mt = 0; mt < 2; ++mt)
#pragma unroll
            for (int nt = 0; nt < 4; ++nt) acc[mt][nt] = vzero8();

        for (int k0 = 0; k0 < F_DIM; k0 += 32) {
            v16bf A0 = load_a_lds(Hbuf, F_DIM, k0, lane, 0);
            v16bf A1 = load_a_lds(Hbuf, F_DIM, k0, lane, 16);
#pragma unroll
            for (int nt = 0; nt < 4; ++nt) {
                int n = hg + nt * 16 + (lane & 15);
                v16bf B = load_b_glb(w2e + (size_t)n * F_DIM, k0, lane);
                acc[0][nt] = WMMA_BF16(A0, B, acc[0][nt]);
                acc[1][nt] = WMMA_BF16(A1, B, acc[1][nt]);
            }
        }
#pragma unroll
        for (int mt = 0; mt < 2; ++mt) {
            int mbase = mt * 16 + ((lane >> 4) << 3);
#pragma unroll
            for (int nt = 0; nt < 4; ++nt) {
                int col = hg + nt * 16 + (lane & 15);
#pragma unroll
                for (int r = 0; r < 8; ++r) {
                    int m = mbase + r;
                    float v = acc[mt][nt][r] * sW[m];
                    atomicAdd(&out[(size_t)sTok[m] * H_DIM + col], v);
                }
            }
        }
    }
}

// ---------------------------------------------------------------------------
extern "C" void kernel_launch(void* const* d_in, const int* in_sizes, int n_in,
                              void* d_out, int out_size, void* d_ws, size_t ws_size,
                              hipStream_t stream) {
    const float* x    = (const float*)d_in[0];   // [4,4096,1024]
    const float* mod  = (const float*)d_in[1];   // [1024]
    const float* gate = (const float*)d_in[2];   // [8,1024]
    const float* w1   = (const float*)d_in[3];   // [8,1024,2048]
    const float* w3   = (const float*)d_in[4];   // [8,1024,2048]
    const float* w2   = (const float*)d_in[5];   // [8,2048,1024]

    float* out    = (float*)d_out;                       // [T*H]
    float* logits = out + (size_t)T_TOK * H_DIM;         // [T*E]

    // workspace layout (~135.3 MB total)
    char* ws = (char*)d_ws;
    int*            counts = (int*)ws;                                     // 256 B
    int*            tlist  = (int*)(ws + 256);                             // E*T*4
    float*          twgt   = (float*)(ws + 256 + (size_t)N_EXP * T_TOK * 4);
    unsigned short* xb     = (unsigned short*)(ws + 256 + (size_t)N_EXP * T_TOK * 8);
    unsigned short* w1t    = xb  + (size_t)T_TOK * H_DIM;
    unsigned short* w3t    = w1t + (size_t)N_EXP * H_DIM * F_DIM;
    unsigned short* w2t    = w3t + (size_t)N_EXP * H_DIM * F_DIM;

    (void)hipMemsetAsync(d_out, 0, (size_t)T_TOK * H_DIM * sizeof(float), stream);
    (void)hipMemsetAsync(counts, 0, 256, stream);

    int n4 = T_TOK * H_DIM / 4;
    cvt_x_kernel<<<n4 / 256, 256, 0, stream>>>(x, xb, n4);
    transpose_cvt_kernel<<<dim3(F_DIM / 32, H_DIM / 32, N_EXP), dim3(32, 8), 0, stream>>>(
        w1, w1t, H_DIM, F_DIM);                          // [H,F] -> [F,H]
    transpose_cvt_kernel<<<dim3(F_DIM / 32, H_DIM / 32, N_EXP), dim3(32, 8), 0, stream>>>(
        w3, w3t, H_DIM, F_DIM);                          // [H,F] -> [F,H]
    transpose_cvt_kernel<<<dim3(H_DIM / 32, F_DIM / 32, N_EXP), dim3(32, 8), 0, stream>>>(
        w2, w2t, F_DIM, H_DIM);                          // [F,H] -> [H,F]

    route_kernel<<<T_TOK / 8, 256, 0, stream>>>(x, mod, gate, logits,
                                                counts, tlist, twgt);

    size_t smem = (size_t)(TILE_M * H_DIM + TILE_M * F_DIM) * sizeof(unsigned short)
                + TILE_M * sizeof(int) + TILE_M * sizeof(float);
    moe_mlp_kernel<<<N_EXP * TILES_PER_EXP, 256, smem, stream>>>(
        xb, w1t, w3t, w2t, counts, tlist, twgt, out);
}
